// Router_75694503624800
// MI455X (gfx1250) — matrix operand order repaired and checked by reading, not compiled, not run on hardware
//
#include <hip/hip_runtime.h>
#include <hip/hip_bf16.h>
#include <math.h>

// ---------------------------------------------------------------------------
// CDNA5 (gfx1250) wave32 WMMA implementation of the transformer-router ref.
// GEMMs: v_wmma_f32_16x16x32_f16, LDS tiles staged by the Tensor Data Mover
// (tensor_load_to_lds + s_wait_tensorcnt) with padded LDS rows.
// softmax / layernorm / residuals stay fp32.
// ---------------------------------------------------------------------------

typedef __attribute__((ext_vector_type(16))) _Float16 v16h;
typedef __attribute__((ext_vector_type(8)))  float    v8f;
typedef __attribute__((ext_vector_type(4)))  unsigned int u32x4;
typedef __attribute__((ext_vector_type(8)))  int      i32x8;
typedef __attribute__((ext_vector_type(4)))  int      i32x4;

#define WMMA16(a, b, c) \
  __builtin_amdgcn_wmma_f32_16x16x32_f16(false, (a), false, (b), (short)0, (c), false, false)

#define Dm      256
#define Hh      8
#define HDm     32
#define Bm      64
#define LEN     512
#define TOK     (Bm * LEN)      // 32768 padded tokens
#define LRELU   0.01f

#if defined(__has_builtin)
# if __has_builtin(__builtin_amdgcn_tensor_load_to_lds)
#  define HAVE_TDM 1
# else
#  define HAVE_TDM 0
# endif
#else
# define HAVE_TDM 0
#endif

static __device__ __forceinline__ void wait_tensorcnt0() {
#if __has_builtin(__builtin_amdgcn_s_wait_tensorcnt)
  __builtin_amdgcn_s_wait_tensorcnt((short)0);
#else
  asm volatile("s_wait_tensorcnt 0x0" ::: "memory");
#endif
}

// TDM 2-D tile load: tile_h rows x tile_w fp32 elements, row stride = K elems,
// into LDS at lds_off with 4-dword padding after every 32 dwords (LDS row
// stride = 36 dwords). D# bit layout per CDNA5 ISA ch.8.
static __device__ __forceinline__ void tdm_load_2d(unsigned lds_off,
                                                   const void* gptr,
                                                   unsigned k_stride,
                                                   unsigned tile_w,
                                                   unsigned tile_h) {
  unsigned long long ga = (unsigned long long)gptr;
  u32x4 g0;
  g0[0] = 1u;                                   // count=1 (valid user D#)
  g0[1] = lds_off;                              // LDS byte address
  g0[2] = (unsigned)ga;                         // global_addr[31:0]
  g0[3] = (unsigned)(ga >> 32) | 0x80000000u;   // addr[56:32] | type=2<<30
  unsigned td0 = k_stride;                      // tensor_dim0 (elements)
  unsigned td1 = 0x00100000u;                   // tensor_dim1: plenty of rows
  i32x8 g1;
  g1[0] = (int)((2u << 16)        // data_size = 4 bytes
              | (1u << 20)        // pad_enable
              | (4u << 22)        // pad_interval: 32 dwords
              | (3u << 25));      // pad_amount: 4 dwords
  g1[1] = (int)((td0 & 0xFFFFu) << 16);                          // dim0[15:0]
  g1[2] = (int)((td0 >> 16) | ((td1 & 0xFFFFu) << 16));          // dim0 hi|dim1 lo
  g1[3] = (int)((td1 >> 16) | (tile_w << 16));                   // dim1 hi|tile0
  g1[4] = (int)(tile_h & 0xFFFFu);                               // tile1 (tile2=0)
  g1[5] = (int)k_stride;                                         // dim0_stride lo
  g1[6] = 0;                                                     // stride hi
  g1[7] = 0;
  i32x4 z4 = {0, 0, 0, 0};
#if HAVE_TDM
# if defined(__clang_major__) && __clang_major__ >= 23
  i32x8 z8 = {0, 0, 0, 0, 0, 0, 0, 0};
  __builtin_amdgcn_tensor_load_to_lds(g0, g1, z4, z4, z8, 0);
# else
  __builtin_amdgcn_tensor_load_to_lds(g0, g1, z4, z4, 0);
# endif
#else
  (void)g0; (void)g1; (void)z4;
#endif
}

// ---- fragment gathers (contiguous k = 16*(lane/16) + s order, same for A and B) ----------
// fp32-LDS variants: convert to f16 at read (v_cvt_pk_f16_f32 co-issues).
static __device__ __forceinline__ v16h frag_a_f32(const float* base, int stride,
                                                  int row0, int lane) {
  const float* p = base + (row0 + (lane & 15)) * stride;
  int g = lane >> 4;
  v16h r;
#pragma unroll
  for (int v = 0; v < 8; ++v) {
    int k = (v << 1) + (g << 4);
    r[2 * v]     = (_Float16)p[k];
    r[2 * v + 1] = (_Float16)p[k + 1];
  }
  return r;
}
static __device__ __forceinline__ v16h frag_b_f32(const float* base, int stride,
                                                  int col0, int lane) {
  const float* p = base + (col0 + (lane & 15)) * stride;
  int g = lane >> 4;
  v16h r;
#pragma unroll
  for (int v = 0; v < 8; ++v) {
    int k = (v << 1) + (g << 4);
    r[2 * v]     = (_Float16)p[k];
    r[2 * v + 1] = (_Float16)p[k + 1];
  }
  return r;
}
// f16-LDS A-fragment (attention P-bounce)
static __device__ __forceinline__ v16h frag_a_lds(const _Float16* base, int stride,
                                                  int row0, int lane) {
  const _Float16* p = base + (row0 + (lane & 15)) * stride;
  int g = lane >> 4;
  v16h r;
#pragma unroll
  for (int v = 0; v < 8; ++v) {
    int k = (v << 1) + (g << 4);
    r[2 * v]     = p[k];
    r[2 * v + 1] = p[k + 1];
  }
  return r;
}

// ---------------------------------------------------------------------------
// Generic WMMA GEMM: C[M,N] = act(A[M,K] @ W[N,K]^T [+ bias] [+ resid])
// M,N multiples of 64; K multiple of 32 (true for all call sites).
// Block = 128 threads (4 waves); block tile 64x64; wave tile 32x32.
// Double-buffered LDS tiles filled by the TDM: wave0 issues the next tile's
// tensor_load_to_lds, all waves run WMMAs on the current tile, wave0 waits
// TENSORcnt, barrier, swap. ACT: 0 none, 1 relu, 2 leaky(0.01).
// ---------------------------------------------------------------------------
#define GSTR 36   // fp32 LDS row stride: 32 dwords + 4 pad (matches TDM pad)

template <int ACT, bool BIAS, bool RESID>
__global__ __launch_bounds__(128) void gemm_wmma(
    const float* __restrict__ A, const float* __restrict__ W,
    const float* __restrict__ bias, const float* __restrict__ resid,
    float* __restrict__ C, int N, int K, int ldc) {
  __shared__ float As[2][64 * GSTR];
  __shared__ float Ws[2][64 * GSTR];

  const int bm = blockIdx.y * 64;
  const int bn = blockIdx.x * 64;
  const int tid  = threadIdx.x;
  const int lane = tid & 31;
  const int wave = tid >> 5;
  const int wr = (wave >> 1) * 32;   // wave tile row in block tile
  const int wc = (wave & 1) * 32;    // wave tile col in block tile

  auto stage = [&](int buf, int k0) {
#if HAVE_TDM
    if (wave == 0) {
      tdm_load_2d((unsigned)(size_t)&As[buf][0], A + (size_t)bm * K + k0, K, 32, 64);
      tdm_load_2d((unsigned)(size_t)&Ws[buf][0], W + (size_t)bn * K + k0, K, 32, 64);
    }
#else
#pragma unroll
    for (int it = 0; it < 4; ++it) {
      int i = tid + 128 * it;
      int r = i >> 3, c = (i & 7) << 2;
      *(float4*)&As[buf][r * GSTR + c] = *(const float4*)&A[(size_t)(bm + r) * K + k0 + c];
      *(float4*)&Ws[buf][r * GSTR + c] = *(const float4*)&W[(size_t)(bn + r) * K + k0 + c];
    }
#endif
  };
  auto stage_wait = [&]() {
#if HAVE_TDM
    if (wave == 0) wait_tensorcnt0();
#endif
  };

  v8f c00 = {}, c01 = {}, c10 = {}, c11 = {};

  stage(0, 0);
  stage_wait();
  __syncthreads();

  int buf = 0;
  for (int k0 = 0; k0 < K; k0 += 32) {
    const bool has_next = (k0 + 32) < K;
    if (has_next) stage(buf ^ 1, k0 + 32);   // async DMA into other buffer

    v16h a0 = frag_a_f32(As[buf], GSTR, wr,      lane);
    v16h a1 = frag_a_f32(As[buf], GSTR, wr + 16, lane);
    v16h b0 = frag_b_f32(Ws[buf], GSTR, wc,      lane);
    v16h b1 = frag_b_f32(Ws[buf], GSTR, wc + 16, lane);

    c00 = WMMA16(a0, b0, c00);
    c01 = WMMA16(a0, b1, c01);
    c10 = WMMA16(a1, b0, c10);
    c11 = WMMA16(a1, b1, c11);

    if (has_next) stage_wait();              // TENSORcnt -> 0 on wave0
    __syncthreads();
    buf ^= 1;
  }

  // epilogue: C/D layout -> lanes 0-15: M=r, lanes 16-31: M=r+8
  const int g = lane >> 4, ln = lane & 15;
#pragma unroll
  for (int i = 0; i < 2; ++i)
#pragma unroll
    for (int j = 0; j < 2; ++j) {
      const v8f& acc = i == 0 ? (j == 0 ? c00 : c01) : (j == 0 ? c10 : c11);
      const int col = bn + wc + j * 16 + ln;
      const float bv = BIAS ? bias[col] : 0.f;
#pragma unroll
      for (int r = 0; r < 8; ++r) {
        int row = bm + wr + i * 16 + r + g * 8;
        float v = acc[r] + bv;
        if (RESID) v += resid[(size_t)row * ldc + col];
        if (ACT == 1)      v = v > 0.f ? v : 0.f;
        else if (ACT == 2) v = v > 0.f ? v : LRELU * v;
        C[(size_t)row * ldc + col] = v;
      }
    }
  (void)N;
}

// ---------------------------------------------------------------------------
// Flash-style attention: one workgroup per (b,h); 8 waves; each wave owns
// four 16-row Q blocks. Q/K/V fragments read from global fp32 (QKV fits in
// the 192MB L2) with inline f16 conversion; P bounced through per-wave LDS.
// ---------------------------------------------------------------------------
#define PSTR 36

__global__ __launch_bounds__(256) void attn_flash(
    const float* __restrict__ QKV, const int* __restrict__ counts,
    float* __restrict__ O) {
  __shared__ _Float16 Pb[8][16 * PSTR];

  const int b = blockIdx.x >> 3;
  const int h = blockIdx.x & 7;
  const int tid = threadIdx.x, lane = tid & 31, wave = tid >> 5;
  const int g = lane >> 4, ln = lane & 15;
  const int cnt = counts[b];
  const float* base = QKV + (size_t)b * LEN * 768;
  const float scale = 0.17677669529663687f;  // 1/sqrt(32)

  for (int rb = wave; rb < 32; rb += 8) {
    const int qrow = rb * 16;
    // Q fragment (A layout) straight from global
    v16h qf;
    {
      const float* qp = base + (size_t)(qrow + ln) * 768 + h * HDm;
#pragma unroll
      for (int v = 0; v < 8; ++v) {
        int k = (v << 1) + (g << 4);
        qf[2 * v]     = (_Float16)qp[k];
        qf[2 * v + 1] = (_Float16)qp[k + 1];
      }
    }
    v8f o0 = {}, o1 = {};
    float mr[8], lr[8];
#pragma unroll
    for (int r = 0; r < 8; ++r) { mr[r] = -INFINITY; lr[r] = 0.f; }

    for (int j = 0; j < LEN; j += 32) {
      // K^T fragments (B layout): key index = column
      v16h kf0, kf1;
      {
        const float* k0p = base + (size_t)(j + ln) * 768 + 256 + h * HDm;
        const float* k1p = base + (size_t)(j + 16 + ln) * 768 + 256 + h * HDm;
        if (j + 32 < LEN) {   // keep next chunk warm in L2/WGP$
          __builtin_prefetch(k0p + 32 * 768, 0, 0);
          __builtin_prefetch(k0p + 32 * 768 + 256, 0, 0);
        }
#pragma unroll
        for (int v = 0; v < 8; ++v) {
          int k = (v << 1) + (g << 4);
          kf0[2 * v]     = (_Float16)k0p[k];
          kf0[2 * v + 1] = (_Float16)k0p[k + 1];
          kf1[2 * v]     = (_Float16)k1p[k];
          kf1[2 * v + 1] = (_Float16)k1p[k + 1];
        }
      }
      v8f zero = {};
      v8f s0 = WMMA16(qf, kf0, zero);
      v8f s1 = WMMA16(qf, kf1, zero);

      const bool k0in = (j + ln) < cnt;
      const bool k1in = (j + 16 + ln) < cnt;
#pragma unroll
      for (int r = 0; r < 8; ++r) {
        float x0 = k0in ? s0[r] * scale : -1.0e9f;
        float x1 = k1in ? s1[r] * scale : -1.0e9f;
        float rm = fmaxf(x0, x1);
#pragma unroll
        for (int off = 1; off < 16; off <<= 1)
          rm = fmaxf(rm, __shfl_xor(rm, off, 16));
        float mn   = fmaxf(mr[r], rm);
        float corr = __expf(mr[r] - mn);
        float p0   = __expf(x0 - mn);
        float p1   = __expf(x1 - mn);
        float rs   = p0 + p1;
#pragma unroll
        for (int off = 1; off < 16; off <<= 1)
          rs += __shfl_xor(rs, off, 16);
        lr[r] = lr[r] * corr + rs;
        mr[r] = mn;
        o0[r] *= corr; o1[r] *= corr;
        s0[r] = p0;    s1[r] = p1;
      }
      // C-layout -> A-layout bounce of P (wave-private, no barrier needed)
      _Float16* P = Pb[wave];
#pragma unroll
      for (int r = 0; r < 8; ++r) {
        int row = r + g * 8;
        P[row * PSTR + ln]      = (_Float16)s0[r];
        P[row * PSTR + 16 + ln] = (_Float16)s1[r];
      }
      v16h pf = frag_a_lds(P, PSTR, 0, lane);
      // V fragments (B layout but key = row in memory): strided global reads
      v16h vf0, vf1;
#pragma unroll
      for (int v = 0; v < 8; ++v) {
        int k = (v << 1) + (g << 4);
        const float* v0p = base + (size_t)(j + k) * 768 + 512 + h * HDm;
        const float* v1p = base + (size_t)(j + k + 1) * 768 + 512 + h * HDm;
        vf0[2 * v]     = (_Float16)v0p[ln];
        vf0[2 * v + 1] = (_Float16)v1p[ln];
        vf1[2 * v]     = (_Float16)v0p[16 + ln];
        vf1[2 * v + 1] = (_Float16)v1p[16 + ln];
      }
      o0 = WMMA16(pf, vf0, o0);
      o1 = WMMA16(pf, vf1, o1);
    }
#pragma unroll
    for (int r = 0; r < 8; ++r) {
      int row = qrow + r + g * 8;
      float inv = 1.f / lr[r];
      O[((size_t)b * LEN + row) * Dm + h * HDm + ln]      = o0[r] * inv;
      O[((size_t)b * LEN + row) * Dm + h * HDm + 16 + ln] = o1[r] * inv;
    }
  }
}

// ---------------------------------------------------------------------------
// Row-wise LayerNorm, one wave per row
// ---------------------------------------------------------------------------
__global__ void ln_rows(const float* __restrict__ X, const float* __restrict__ sc,
                        const float* __restrict__ bi, float* __restrict__ Y,
                        int rows, int W) {
  int row  = blockIdx.x * (blockDim.x >> 5) + (threadIdx.x >> 5);
  int lane = threadIdx.x & 31;
  if (row >= rows) return;
  const float* x = X + (size_t)row * W;
  float s = 0.f, sq = 0.f;
  for (int i = lane; i < W; i += 32) { float v = x[i]; s += v; sq += v * v; }
#pragma unroll
  for (int off = 1; off < 32; off <<= 1) {
    s  += __shfl_xor(s, off, 32);
    sq += __shfl_xor(sq, off, 32);
  }
  float mean = s / W;
  float var  = sq / W - mean * mean;
  float rstd = rsqrtf(var + 1e-5f);
  float* y = Y + (size_t)row * W;
  for (int i = lane; i < W; i += 32)
    y[i] = (x[i] - mean) * rstd * sc[i] + bi[i];
}

// ---------------------------------------------------------------------------
// small helpers
// ---------------------------------------------------------------------------
__global__ void zero_f32(float* p, size_t n) {
  size_t i = (size_t)blockIdx.x * blockDim.x + threadIdx.x;
  if (i < n) p[i] = 0.f;
}
__global__ void zero_counts(int* counts) {
  if (threadIdx.x < Bm) counts[threadIdx.x] = 0;
}
__global__ void count_bids(const int* __restrict__ bid, int* counts, int n) {
  int i = blockIdx.x * blockDim.x + threadIdx.x;
  if (i < n) atomicAdd(&counts[bid[i]], 1);
}
__global__ void scan_counts(const int* __restrict__ counts, int* starts) {
  if (threadIdx.x == 0) {
    int acc = 0;
    for (int i = 0; i < Bm; ++i) { starts[i] = acc; acc += counts[i]; }
  }
}
__global__ void scatter_pad(const float* __restrict__ node, const int* __restrict__ bid,
                            const int* __restrict__ starts, float* __restrict__ X) {
  int t = blockIdx.x;            // token
  int d = threadIdx.x;           // feature
  int b = bid[t];
  int pos = t - starts[b];
  X[((size_t)b * LEN + pos) * Dm + d] = node[(size_t)t * Dm + d];
}
__global__ void copy_node0(const float* __restrict__ X, float* __restrict__ F) {
  int b = blockIdx.x, d = threadIdx.x;
  F[(size_t)b * 512 + d] = X[((size_t)b * LEN) * Dm + d];
}
__global__ void head_small(const float* __restrict__ feat, const float* __restrict__ W,
                           const float* __restrict__ bias, float* __restrict__ out) {
  int t = threadIdx.x;
  if (t >= Bm * 5) return;
  int r = t / 5, c = t % 5;
  float acc = bias[c];
  const float* f = feat + (size_t)r * Dm;
  const float* w = W + (size_t)c * Dm;
  for (int k = 0; k < Dm; ++k) acc += f[k] * w[k];
  out[(size_t)r * 5 + c] = acc;
}

// ---------------------------------------------------------------------------
// host driver
// ---------------------------------------------------------------------------
extern "C" void kernel_launch(void* const* d_in, const int* in_sizes, int n_in,
                              void* d_out, int out_size, void* d_ws, size_t ws_size,
                              hipStream_t stream) {
  (void)n_in; (void)out_size; (void)ws_size;
  const float* node = (const float*)d_in[0];
  const float* img  = (const float*)d_in[2];
  const int*   bid  = (const int*)d_in[3];
  const int Ntok = in_sizes[0] / Dm;

  // params: 4 + 12*layer ... then tail
  auto P = [&](int i) { return (const float*)d_in[i]; };
  const int LP = 4;                 // first layer param index
  const int TP = LP + 12 * 4;       // tail params start (fc_w)

  // workspace layout (fp32)
  float* X    = (float*)d_ws;
  float* X2   = X    + (size_t)TOK * Dm;
  float* QKV  = X2   + (size_t)TOK * Dm;
  float* HIDb = QKV  + (size_t)TOK * 768;
  float* ATT  = HIDb;                       // reuse FFN-hidden buffer
  float* F0   = HIDb + (size_t)TOK * 1024;  // concat [node0 | img_fc] 64x512
  float* F0n  = F0   + Bm * 512;
  float* H1   = F0n  + Bm * 512;
  float* H1n  = H1   + Bm * 1024;
  float* F2   = H1n  + Bm * 1024;
  float* F2n  = F2   + Bm * 512;
  int* counts = (int*)(F2n + Bm * 512);
  int* starts = counts + Bm;

  float* outF = (float*)d_out;              // features 64x256
  float* outL = outF + Bm * Dm;             // logits   64x5

  // ---- pad / scatter ----
  zero_counts<<<1, 64, 0, stream>>>(counts);
  count_bids<<<(Ntok + 255) / 256, 256, 0, stream>>>(bid, counts, Ntok);
  scan_counts<<<1, 32, 0, stream>>>(counts, starts);
  {
    size_t n = (size_t)TOK * Dm;
    zero_f32<<<(unsigned)((n + 255) / 256), 256, 0, stream>>>(X, n);
  }
  scatter_pad<<<Ntok, Dm, 0, stream>>>(node, bid, starts, X);

  // ---- encoder layers ----
  for (int l = 0; l < 4; ++l) {
    const int p = LP + 12 * l;
    const float* qkv_w  = P(p + 0);  const float* qkv_b  = P(p + 1);
    const float* out_w  = P(p + 2);  const float* out_b  = P(p + 3);
    const float* lin1_w = P(p + 4);  const float* lin1_b = P(p + 5);
    const float* lin2_w = P(p + 6);  const float* lin2_b = P(p + 7);
    const float* n1_s   = P(p + 8);  const float* n1_b   = P(p + 9);
    const float* n2_s   = P(p + 10); const float* n2_b   = P(p + 11);

    gemm_wmma<0, true, false><<<dim3(768 / 64, TOK / 64), 128, 0, stream>>>(
        X, qkv_w, qkv_b, nullptr, QKV, 768, Dm, 768);
    attn_flash<<<Bm * Hh, 256, 0, stream>>>(QKV, counts, ATT);
    gemm_wmma<0, true, true><<<dim3(Dm / 64, TOK / 64), 128, 0, stream>>>(
        ATT, out_w, out_b, X, X2, Dm, Dm, Dm);
    ln_rows<<<TOK / 8, 256, 0, stream>>>(X2, n1_s, n1_b, X, TOK, Dm);
    gemm_wmma<1, true, false><<<dim3(1024 / 64, TOK / 64), 128, 0, stream>>>(
        X, lin1_w, lin1_b, nullptr, HIDb, 1024, Dm, 1024);
    gemm_wmma<0, true, true><<<dim3(Dm / 64, TOK / 64), 128, 0, stream>>>(
        HIDb, lin2_w, lin2_b, X, X2, Dm, 1024, Dm);
    ln_rows<<<TOK / 8, 256, 0, stream>>>(X2, n2_s, n2_b, X, TOK, Dm);
  }

  // ---- head ----
  copy_node0<<<Bm, Dm, 0, stream>>>(X, F0);                       // F0[:, :256]
  gemm_wmma<0, true, false><<<dim3(Dm / 64, Bm / 64), 128, 0, stream>>>(
      img, P(TP + 0), P(TP + 1), nullptr, F0 + 256, Dm, 3072, 512);

  ln_rows<<<Bm / 8, 256, 0, stream>>>(F0, P(TP + 4), P(TP + 5), F0n, Bm, 512);
  gemm_wmma<2, true, false><<<dim3(1024 / 64, Bm / 64), 128, 0, stream>>>(
      F0n, P(TP + 6), P(TP + 7), nullptr, H1, 1024, 512, 1024);

  ln_rows<<<Bm / 8, 256, 0, stream>>>(H1, P(TP + 8), P(TP + 9), H1n, Bm, 1024);
  gemm_wmma<2, true, false><<<dim3(512 / 64, Bm / 64), 128, 0, stream>>>(
      H1n, P(TP + 10), P(TP + 11), nullptr, F2, 512, 1024, 512);

  ln_rows<<<Bm / 8, 256, 0, stream>>>(F2, P(TP + 12), P(TP + 13), F2n, Bm, 512);
  gemm_wmma<2, true, false><<<dim3(Dm / 64, Bm / 64), 128, 0, stream>>>(
      F2n, P(TP + 14), P(TP + 15), nullptr, outF, Dm, 512, Dm);

  head_small<<<1, 320, 0, stream>>>(outF, P(TP + 2), P(TP + 3), outL);
}
